// TS_85676007621254
// MI455X (gfx1250) — matrix operand order repaired and checked by reading, not compiled, not run on hardware
//
#include <hip/hip_runtime.h>
#include <stdint.h>

typedef __attribute__((ext_vector_type(16))) __bf16 v16bf;
typedef __attribute__((ext_vector_type(8)))  float  v8f;

union Frag {
    v16bf v;
    uint4 q[2];
};

#define WMMA_BF16(c, a, b) \
    __builtin_amdgcn_wmma_f32_16x16x32_bf16(false, (a), false, (b), (short)0, (c), false, false)

__device__ __forceinline__ uint16_t f2bf(float x) {
    uint32_t u = __float_as_uint(x);
    return (uint16_t)((u + 0x7FFFu + ((u >> 16) & 1u)) >> 16);
}
__device__ __forceinline__ uint32_t pack2bf(float a, float b) {
    return (uint32_t)f2bf(a) | ((uint32_t)f2bf(b) << 16);
}

// ---------------------------------------------------------------------------
// A-fragment (16x32 bf16) from row-major bf16 with K/8 uint4's per row.
// lane: M = lane&15 ; dwords kstep*16 + (lane<16?0:4) .. (+3) and +8..+11
// ---------------------------------------------------------------------------
__device__ __forceinline__ v16bf load_afrag(const uint4* A4, int row, int Kd8,
                                            int ks, int lane) {
    Frag f;
    size_t base = (size_t)row * Kd8 + ks * 4 + ((lane >> 4) & 1);
    f.q[0] = A4[base];
    f.q[1] = A4[base + 2];
    return f.v;
}

// B-fragment (32x16 bf16) from pre-packed layout: frag = ks*Ntiles+nt,
// each frag is 32 lanes x 8 dwords contiguous.
__device__ __forceinline__ v16bf load_bfrag(const uint4* Bp, int Ntiles,
                                            int nt, int ks, int lane) {
    Frag f;
    size_t base = (((size_t)(ks * Ntiles + nt) * 32) + lane) * 2;
    f.q[0] = Bp[base];
    f.q[1] = Bp[base + 1];
    return f.v;
}

// ---------------------------------------------------------------------------
// Pre-pack a row-major f32 weight [K,N] into bf16 WMMA B-fragments.
// ---------------------------------------------------------------------------
__global__ void prepack_w(const float* __restrict__ W, uint32_t* __restrict__ dst,
                          int K, int N) {
    int Ntiles = N >> 4;
    int total = (K * N) >> 1;
    for (int idx = blockIdx.x * blockDim.x + threadIdx.x; idx < total;
         idx += gridDim.x * blockDim.x) {
        int v = idx & 7;
        int lane = (idx >> 3) & 31;
        int f = idx >> 8;
        int nt = f % Ntiles;
        int ks = f / Ntiles;
        int k = ks * 32 + ((v & 4) ? 16 : 0) + ((lane & 16) ? 8 : 0) + 2 * (v & 3);
        int n = nt * 16 + (lane & 15);
        dst[idx] = pack2bf(W[(size_t)k * N + n], W[(size_t)(k + 1) * N + n]);
    }
}

__global__ void cvt_bf16(const float* __restrict__ src, uint16_t* __restrict__ dst,
                         size_t n) {
    for (size_t i = blockIdx.x * (size_t)blockDim.x + threadIdx.x; i < n;
         i += (size_t)gridDim.x * blockDim.x)
        dst[i] = f2bf(src[i]);
}

__global__ void tanh_bf16(const float* __restrict__ src, uint16_t* __restrict__ dst,
                          size_t n) {
    for (size_t i = blockIdx.x * (size_t)blockDim.x + threadIdx.x; i < n;
         i += (size_t)gridDim.x * blockDim.x)
        dst[i] = f2bf(tanhf(src[i]));
}

// ---------------------------------------------------------------------------
// Generic bf16 WMMA GEMM: C[M,N] = act(A[M,K] @ W[K,N] + bias)
// Each wave computes a 16 x 64 block (1 m-tile x 4 n-tiles): the A-fragment
// is loaded once per k-step and feeds 4 WMMAs (4x arithmetic intensity on A).
// Requires N % 64 == 0 (true for all call sites: 128/256/512).
// ---------------------------------------------------------------------------
__global__ __launch_bounds__(256) void gemm_bf16(
    const uint4* __restrict__ A4, const uint4* __restrict__ Bp,
    const float* __restrict__ bias, float* __restrict__ Cf,
    uint16_t* __restrict__ Cbf, int M, int N, int K, int act) {
    int lane = threadIdx.x & 31;
    int wv = threadIdx.x >> 5;
    int Ntiles = N >> 4, Mtiles = M >> 4, Ksteps = K >> 5, Kd8 = K >> 3;
    int Ngroups = Ntiles >> 2;
    int nT = Mtiles * Ngroups;
    int gw = blockIdx.x * 8 + wv;
    int stride = gridDim.x * 8;
    for (int t = gw; t < nT; t += stride) {
        int mt = t / Ngroups, ng = t % Ngroups;
        int nt0 = ng * 4;
        int arow = mt * 16 + (lane & 15);
        v8f c[4];
#pragma unroll
        for (int j = 0; j < 4; ++j) c[j] = (v8f){};
        for (int ks = 0; ks < Ksteps; ++ks) {
            __builtin_prefetch(
                Bp + (((size_t)((ks + 1) * Ntiles + nt0) * 32) + lane) * 2, 0, 0);
            v16bf a = load_afrag(A4, arow, Kd8, ks, lane);
#pragma unroll
            for (int j = 0; j < 4; ++j) {
                v16bf b = load_bfrag(Bp, Ntiles, nt0 + j, ks, lane);
                c[j] = WMMA_BF16(c[j], a, b);
            }
        }
        int rb = mt * 16 + ((lane & 16) ? 8 : 0);
#pragma unroll
        for (int j = 0; j < 4; ++j) {
            int col = (nt0 + j) * 16 + (lane & 15);
            float bv = bias ? bias[col] : 0.f;
#pragma unroll
            for (int r = 0; r < 8; ++r) {
                float v = c[j][r] + bv;
                if (act == 1) v = fmaxf(v, 0.f);
                size_t idx = (size_t)(rb + r) * N + col;
                Cf[idx] = v;
                if (Cbf) Cbf[idx] = f2bf(v);
            }
        }
    }
}

// ---------------------------------------------------------------------------
// Block-wide LayerNorm over buf[0..n), gamma/beta indexed [row*C+col].
// ---------------------------------------------------------------------------
__device__ void block_layernorm(float* buf, int n, int C, const float* g,
                                const float* bt, float* red, int tid) {
    float s = 0.f, s2 = 0.f;
    for (int i = tid; i < n; i += 256) {
        float v = buf[i];
        s += v;
        s2 += v * v;
    }
    for (int o = 16; o > 0; o >>= 1) {
        s += __shfl_xor(s, o, 32);
        s2 += __shfl_xor(s2, o, 32);
    }
    int lane = tid & 31, wv = tid >> 5;
    if (lane == 0) { red[wv] = s; red[8 + wv] = s2; }
    __syncthreads();
    if (tid == 0) {
        float a = 0.f, b = 0.f;
        for (int i = 0; i < 8; ++i) { a += red[i]; b += red[8 + i]; }
        float mu = a / n;
        float var = b / n - mu * mu;
        red[16] = mu;
        red[17] = rsqrtf(var + 1e-5f);
    }
    __syncthreads();
    float mu = red[16], rs = red[17];
    for (int i = tid; i < n; i += 256) {
        int row = i / C, col = i % C;
        buf[i] = (buf[i] - mu) * rs * g[row * C + col] + bt[row * C + col];
    }
    __syncthreads();
}

// ---------------------------------------------------------------------------
// Fused RMC step: one workgroup per batch element (see round-1 analysis).
// qkv GEMM keeps 6 accumulators live so the LDS A-fragment is loaded once
// per k-step for 6 WMMAs; MLP GEMMs keep 2.
// ---------------------------------------------------------------------------
__global__ __launch_bounds__(256) void rmc_step(
    const float* __restrict__ memIn, const float* __restrict__ inpF,
    const uint4* __restrict__ Wqkv_p, const float* __restrict__ bqkv,
    const float* __restrict__ lnqg, const float* __restrict__ lnqb,
    const uint4* __restrict__ Wmlp_p, const float* __restrict__ bmlp,
    const float* __restrict__ ln1g, const float* __restrict__ ln1b,
    const float* __restrict__ ln2g, const float* __restrict__ ln2b,
    const float* __restrict__ gateIn, const float* __restrict__ gateMem,
    const float* __restrict__ fbias, const float* __restrict__ ibias,
    float* __restrict__ memOut, uint16_t* __restrict__ memOutBf,
    uint16_t* __restrict__ tanhBf) {
    __shared__ __align__(16) uint32_t s_mbf[16 * 128];   // 16x256 bf16 (A tile)
    __shared__ __align__(16) uint32_t s_abf[16 * 128];   // 16x256 bf16 (MLP A)
    __shared__ float s_mf[9 * 256];                      // m (f32, residual)
    __shared__ float s_qkv[9 * 768];                     // qkv f32 / reused as a2
    __shared__ float s_m1[9 * 256];                      // m1 f32
    __shared__ float s_w[4 * 81];                        // attention weights
    __shared__ float s_red[32];

    int tid = threadIdx.x, lane = tid & 31, wv = tid >> 5;
    int bIdx = blockIdx.x;
    uint16_t* mbf16 = (uint16_t*)s_mbf;
    uint16_t* abf16 = (uint16_t*)s_abf;

    // Build m = [mem ; inp], pad rows 9..15 with zero; keep f32 + bf16 copies.
    for (int e = tid; e < 4096; e += 256) {
        int row = e >> 8, col = e & 255;
        float v = 0.f;
        if (row < 8)       v = memIn[((size_t)bIdx * 8 + row) * 256 + col];
        else if (row == 8) v = inpF[(size_t)bIdx * 256 + col];
        mbf16[e] = f2bf(v);
        if (row < 9) s_mf[e] = v;
    }
    for (int e = tid; e < 2048; e += 256) s_abf[e] = 0u;
    __syncthreads();

    // qkv GEMM: [16,256] @ [256,768]; wave wv owns n-tiles wv, wv+8, ..., wv+40
    const uint4* mb4 = (const uint4*)s_mbf;
    {
        v8f acc[6];
#pragma unroll
        for (int i = 0; i < 6; ++i) acc[i] = (v8f){};
        for (int ks = 0; ks < 8; ++ks) {
            v16bf a = load_afrag(mb4, lane & 15, 32, ks, lane);
#pragma unroll
            for (int i = 0; i < 6; ++i) {
                v16bf b = load_bfrag(Wqkv_p, 48, wv + 8 * i, ks, lane);
                acc[i] = WMMA_BF16(acc[i], a, b);
            }
        }
        int rb = (lane & 16) ? 8 : 0;
#pragma unroll
        for (int i = 0; i < 6; ++i) {
            int col = (wv + 8 * i) * 16 + (lane & 15);
            float bv = bqkv[col];
#pragma unroll
            for (int r = 0; r < 8; ++r) {
                int row = rb + r;
                if (row < 9) s_qkv[row * 768 + col] = acc[i][r] + bv;
            }
        }
    }
    __syncthreads();

    block_layernorm(s_qkv, 6912, 768, lnqg, lnqb, s_red, tid);

    // Attention logits + softmax (rows: 4 heads x 9 queries)
    if (tid < 36) {
        int h = tid / 9, qi = tid % 9;
        const float* qr = &s_qkv[qi * 768 + h * 192];
        float lg[9];
        float mx = -1e30f;
        for (int j = 0; j < 9; ++j) {
            const float* kr = &s_qkv[j * 768 + h * 192 + 64];
            float acc = 0.f;
            for (int d = 0; d < 64; ++d) acc += qr[d] * kr[d];
            acc *= 0.125f;  // 1/sqrt(64)
            lg[j] = acc;
            mx = fmaxf(mx, acc);
        }
        float sum = 0.f;
        for (int j = 0; j < 9; ++j) { lg[j] = expf(lg[j] - mx); sum += lg[j]; }
        float inv = 1.f / sum;
        for (int j = 0; j < 9; ++j) s_w[h * 81 + qi * 9 + j] = lg[j] * inv;
    }
    __syncthreads();

    // att @ v, residual add (each thread owns one (head, dim) column)
    {
        int h = tid >> 6, d = tid & 63;
        for (int i = 0; i < 9; ++i) {
            float acc = 0.f;
#pragma unroll
            for (int j = 0; j < 9; ++j)
                acc += s_w[h * 81 + i * 9 + j] * s_qkv[j * 768 + h * 192 + 128 + d];
            int col = h * 64 + d;
            s_m1[i * 256 + col] = s_mf[i * 256 + col] + acc;
        }
    }
    __syncthreads();

    block_layernorm(s_m1, 2304, 256, ln1g, ln1b, s_red, tid);
    for (int e = tid; e < 2304; e += 256) mbf16[e] = f2bf(s_m1[e]);
    __syncthreads();

    // MLP layer 1: relu(m1 @ Wmlp + b) -> bf16
    {
        v8f acc[2];
#pragma unroll
        for (int i = 0; i < 2; ++i) acc[i] = (v8f){};
        for (int ks = 0; ks < 8; ++ks) {
            v16bf a = load_afrag(mb4, lane & 15, 32, ks, lane);
#pragma unroll
            for (int i = 0; i < 2; ++i) {
                v16bf b = load_bfrag(Wmlp_p, 16, wv + 8 * i, ks, lane);
                acc[i] = WMMA_BF16(acc[i], a, b);
            }
        }
        int rb = (lane & 16) ? 8 : 0;
#pragma unroll
        for (int i = 0; i < 2; ++i) {
            int col = (wv + 8 * i) * 16 + (lane & 15);
            float bv = bmlp[col];
#pragma unroll
            for (int r = 0; r < 8; ++r) {
                int row = rb + r;
                if (row < 9) abf16[row * 256 + col] = f2bf(fmaxf(acc[i][r] + bv, 0.f));
            }
        }
    }
    __syncthreads();

    // MLP layer 2 -> f32 (reuse qkv LDS region)
    const uint4* ab4 = (const uint4*)s_abf;
    float* s_af = s_qkv;
    {
        v8f acc[2];
#pragma unroll
        for (int i = 0; i < 2; ++i) acc[i] = (v8f){};
        for (int ks = 0; ks < 8; ++ks) {
            v16bf a = load_afrag(ab4, lane & 15, 32, ks, lane);
#pragma unroll
            for (int i = 0; i < 2; ++i) {
                v16bf b = load_bfrag(Wmlp_p, 16, wv + 8 * i, ks, lane);
                acc[i] = WMMA_BF16(acc[i], a, b);
            }
        }
        int rb = (lane & 16) ? 8 : 0;
#pragma unroll
        for (int i = 0; i < 2; ++i) {
            int col = (wv + 8 * i) * 16 + (lane & 15);
            float bv = bmlp[col];
#pragma unroll
            for (int r = 0; r < 8; ++r) {
                int row = rb + r;
                if (row < 9) s_af[row * 256 + col] = fmaxf(acc[i][r] + bv, 0.f);
            }
        }
    }
    __syncthreads();

    for (int e = tid; e < 2304; e += 256) s_af[e] += s_m1[e];
    __syncthreads();
    block_layernorm(s_af, 2304, 256, ln2g, ln2b, s_red, tid);

    // Gating + state update
    float fb = fbias[0], ib = ibias[0];
    for (int e = tid; e < 2048; e += 256) {
        int s = e >> 8, c = e & 255;
        float mo = s_mf[e];
        float gi_i = gateIn[(size_t)bIdx * 512 + c];
        float gi_f = gateIn[(size_t)bIdx * 512 + 256 + c];
        size_t gmb = ((size_t)bIdx * 8 + s) * 512;
        float gm_i = gateMem[gmb + c];
        float gm_f = gateMem[gmb + 256 + c];
        float ig = 1.f / (1.f + expf(-(gm_i + gi_i + ib)));
        float fg = 1.f / (1.f + expf(-(gm_f + gi_f + fb)));
        float nm = ig * tanhf(s_af[e]) + fg * mo;
        size_t oi = (size_t)bIdx * 2048 + e;
        memOut[oi] = nm;
        memOutBf[oi] = f2bf(nm);
        tanhBf[oi] = f2bf(tanhf(nm));
    }
}

__global__ void logsoftmax128(const float* __restrict__ logits,
                              float* __restrict__ out) {
    int r = blockIdx.x, c = threadIdx.x;
    float x = logits[(size_t)r * 128 + c];
    __shared__ float sh[128];
    sh[c] = x;
    __syncthreads();
    for (int o = 64; o > 0; o >>= 1) {
        if (c < o) sh[c] = fmaxf(sh[c], sh[c + o]);
        __syncthreads();
    }
    float mx = sh[0];
    __syncthreads();
    sh[c] = expf(x - mx);
    __syncthreads();
    for (int o = 64; o > 0; o >>= 1) {
        if (c < o) sh[c] += sh[c + o];
        __syncthreads();
    }
    float lse = logf(sh[0]);
    out[(size_t)r * 128 + c] = x - mx - lse;
}

// ---------------------------------------------------------------------------
extern "C" void kernel_launch(void* const* d_in, const int* in_sizes, int n_in,
                              void* d_out, int out_size, void* d_ws, size_t ws_size,
                              hipStream_t stream) {
    const float* inputs  = (const float*)d_in[0];   // [16,4096,256]
    const float* memory0 = (const float*)d_in[1];   // [4096,8,256]
    const float* W_qkv   = (const float*)d_in[2];
    const float* b_qkv   = (const float*)d_in[3];
    const float* lnqg    = (const float*)d_in[4];
    const float* lnqb    = (const float*)d_in[5];
    const float* W_mlp   = (const float*)d_in[6];
    const float* b_mlp   = (const float*)d_in[7];
    const float* ln1g    = (const float*)d_in[8];
    const float* ln1b    = (const float*)d_in[9];
    const float* ln2g    = (const float*)d_in[10];
    const float* ln2b    = (const float*)d_in[11];
    const float* W_emb   = (const float*)d_in[12];
    const float* b_emb   = (const float*)d_in[13];
    const float* W_ig    = (const float*)d_in[14];
    const float* b_ig    = (const float*)d_in[15];
    const float* W_mg    = (const float*)d_in[16];
    const float* b_mg    = (const float*)d_in[17];
    const float* fbias   = (const float*)d_in[18];
    const float* ibias   = (const float*)d_in[19];
    const float* W_out   = (const float*)d_in[20];
    const float* b_out   = (const float*)d_in[21];

    const int B = 4096, T = 16;

    char* ws = (char*)d_ws;
    size_t off = 0;
    auto carve = [&](size_t bytes) -> char* {
        char* p = ws + off;
        off += (bytes + 255) & ~(size_t)255;
        return p;
    };

    uint32_t* wemb_p = (uint32_t*)carve((size_t)256 * 256 * 2);
    uint32_t* wqkv_p = (uint32_t*)carve((size_t)256 * 768 * 2);
    uint32_t* wmlp_p = (uint32_t*)carve((size_t)256 * 256 * 2);
    uint32_t* wig_p  = (uint32_t*)carve((size_t)256 * 512 * 2);
    uint32_t* wmg_p  = (uint32_t*)carve((size_t)256 * 512 * 2);
    uint32_t* wout_p = (uint32_t*)carve((size_t)2048 * 128 * 2);

    uint16_t* xbf     = (uint16_t*)carve((size_t)T * B * 256 * 2);
    float*    inpF    = (float*)carve((size_t)B * 256 * 4);
    uint16_t* inpBf   = (uint16_t*)carve((size_t)B * 256 * 2);
    float*    gateIn  = (float*)carve((size_t)B * 512 * 4);
    float*    gateMem = (float*)carve((size_t)B * 8 * 512 * 4);
    float*    mem0    = (float*)carve((size_t)B * 2048 * 4);
    float*    mem1    = (float*)carve((size_t)B * 2048 * 4);
    uint16_t* memBf   = (uint16_t*)carve((size_t)B * 2048 * 2);
    uint16_t* tanhBf  = (uint16_t*)carve((size_t)B * 2048 * 2);
    float*    logits  = (float*)carve((size_t)B * 128 * 4);
    float*    memBufs[2] = {mem0, mem1};

    // One-time weight conversion / packing (deterministic; re-run each call)
    prepack_w<<<256, 256, 0, stream>>>(W_emb, wemb_p, 256, 256);
    prepack_w<<<256, 256, 0, stream>>>(W_qkv, wqkv_p, 256, 768);
    prepack_w<<<256, 256, 0, stream>>>(W_mlp, wmlp_p, 256, 256);
    prepack_w<<<256, 256, 0, stream>>>(W_ig,  wig_p,  256, 512);
    prepack_w<<<256, 256, 0, stream>>>(W_mg,  wmg_p,  256, 512);
    prepack_w<<<512, 256, 0, stream>>>(W_out, wout_p, 2048, 128);
    cvt_bf16<<<4096, 256, 0, stream>>>(inputs, xbf, (size_t)T * B * 256);
    tanh_bf16<<<2048, 256, 0, stream>>>(memory0, tanhBf, (size_t)B * 2048);

    for (int t = 0; t < T; ++t) {
        const float* memIn = (t == 0) ? memory0 : memBufs[(t - 1) & 1];
        float* memNext = memBufs[t & 1];

        // inp = x_t @ W_emb + b_emb   (f32 + bf16 outputs)
        gemm_bf16<<<512, 256, 0, stream>>>(
            (const uint4*)(xbf + (size_t)t * B * 256), (const uint4*)wemb_p,
            b_emb, inpF, inpBf, B, 256, 256, 0);
        // gate_in = inp @ W_ig + b_ig
        gemm_bf16<<<512, 256, 0, stream>>>(
            (const uint4*)inpBf, (const uint4*)wig_p, b_ig, gateIn,
            (uint16_t*)nullptr, B, 512, 256, 0);
        // gate_mem = tanh(mem) @ W_mg + b_mg
        gemm_bf16<<<512, 256, 0, stream>>>(
            (const uint4*)tanhBf, (const uint4*)wmg_p, b_mg, gateMem,
            (uint16_t*)nullptr, B * 8, 512, 256, 0);
        // fused attention/MLP/LN/gating step
        rmc_step<<<B, 256, 0, stream>>>(
            memIn, inpF, (const uint4*)wqkv_p, b_qkv, lnqg, lnqb,
            (const uint4*)wmlp_p, b_mlp, ln1g, ln1b, ln2g, ln2b,
            gateIn, gateMem, fbias, ibias, memNext, memBf, tanhBf);
    }

    // logits = mem_final.reshape(B,2048) @ W_out + b_out
    gemm_bf16<<<512, 256, 0, stream>>>(
        (const uint4*)memBf, (const uint4*)wout_p, b_out, logits,
        (uint16_t*)nullptr, B, 128, 2048, 0);
    logsoftmax128<<<B, 128, 0, stream>>>(logits, (float*)d_out);

    // final_mem -> second output region
    hipMemcpyAsync((float*)d_out + (size_t)B * 128, memBufs[(T - 1) & 1],
                   (size_t)B * 2048 * 4, hipMemcpyDeviceToDevice, stream);
}
